// AttentionFusion_77884936945901
// MI455X (gfx1250) — matrix operand order repaired and checked by reading, not compiled, not run on hardware
//
#include <hip/hip_runtime.h>

typedef unsigned int u32;
typedef unsigned short u16;
typedef __bf16 bf16;
typedef __attribute__((ext_vector_type(16))) __bf16 v16bf;
typedef __attribute__((ext_vector_type(8)))  float  v8f;

#if __has_builtin(__builtin_amdgcn_global_load_async_to_lds_b128)
#define USE_ASYNC_LDS 1
// exact param pointee type per hipcc diagnostic: vector_size(16) int
typedef int v4i_ __attribute__((vector_size(16)));
typedef __attribute__((address_space(1))) v4i_ gv4i;   // global (AS1)
typedef __attribute__((address_space(3))) v4i_ lv4i;   // LDS (AS3)
#else
#define USE_ASYNC_LDS 0
#endif

__device__ __forceinline__ void wait_async0() {
#if USE_ASYNC_LDS
#if __has_builtin(__builtin_amdgcn_s_wait_asynccnt)
  __builtin_amdgcn_s_wait_asynccnt(0);
#else
  asm volatile("s_wait_asynccnt 0" ::: "memory");
#endif
#endif
}

union Frag {
  v16bf v;
  uint4 q[2];
  u16   s[16];
};

__device__ __forceinline__ v8f wmma_bf16(const Frag& a, const Frag& b, v8f c) {
  // D = A(16x32) * B(32x16) + C, f32 accumulate
  return __builtin_amdgcn_wmma_f32_16x16x32_bf16(false, a.v, false, b.v,
                                                 (short)0, c, false, false);
}

// ---------------------------------------------------------------------------
// Kernel 1: fused Q/K/V projection.
//   Q[n,c] = (Wq @ Xs + bq) / sqrt(C)  -> Qb [B][N][C] bf16
//   K[n,c] =  Wk @ Xt + bk             -> Kb [B][N][C] bf16
//   V stored transposed                -> Vt [B][C][N] bf16
// WMMA mapping: M = voxel rows, N-dim = c_out, K = c_in.
// ---------------------------------------------------------------------------
__global__ __launch_bounds__(128) void qkv_proj_kernel(
    const float* __restrict__ spat, const float* __restrict__ temp,
    const float* __restrict__ Wq, const float* __restrict__ bq,
    const float* __restrict__ Wk, const float* __restrict__ bk,
    const float* __restrict__ Wv, const float* __restrict__ bv,
    bf16* __restrict__ Qb, bf16* __restrict__ Kb, bf16* __restrict__ Vt)
{
  const int b    = blockIdx.y;
  const int wave = threadIdx.x >> 5;
  const int lane = threadIdx.x & 31;
  const int m    = lane & 15;
  const int half = lane >> 4;
  const int n0   = blockIdx.x * 64 + wave * 16;
  const int nvox = n0 + m;

  // spatial: [B, T, C, H*W]  voxel n = t*1024 + hw ; temporal: [B, C, N]
  const int t  = nvox >> 10;
  const int hw = nvox & 1023;
  const float* spB = spat + ((size_t)(b * 4 + t) * 128) * 1024 + hw;
  const float* tpB = temp + ((size_t)b * 128) * 4096 + nvox;

  // A-fragments of Xs^T and Xt^T (16 rows x 128 k), native cvt to bf16.
  // A-layout (contiguous k, as B): lane(l): v[j]=A[m, ks*32+half*16+j], v[8+j]=A[m, ks*32+half*16+8+j]
  Frag aS[4], aT[4];
#pragma unroll
  for (int ks = 0; ks < 4; ++ks) {
#pragma unroll
    for (int j = 0; j < 8; ++j) {
      const int c1 = ks * 32 + half * 16 + j;
      const int c2 = c1 + 8;
      aS[ks].v[j]     = (bf16)spB[(size_t)c1 * 1024];
      aS[ks].v[8 + j] = (bf16)spB[(size_t)c2 * 1024];
      aT[ks].v[j]     = (bf16)tpB[(size_t)c1 * 4096];
      aT[ks].v[8 + j] = (bf16)tpB[(size_t)c2 * 4096];
    }
  }

  const float qscale = 0.08838834764831845f; // 128^-0.5

#pragma unroll
  for (int co = 0; co < 8; ++co) {
    const int crow = co * 16 + m;  // B-frag column = c_out
    v8f accq = {}, acck = {}, accv = {};
#pragma unroll
    for (int ks = 0; ks < 4; ++ks) {
      const int kb = ks * 32 + half * 16;  // B-layout: 16 contiguous k per lane
      Frag fq, fk, fv;
#pragma unroll
      for (int j = 0; j < 16; ++j) {
        fq.v[j] = (bf16)Wq[(size_t)crow * 128 + kb + j];
        fk.v[j] = (bf16)Wk[(size_t)crow * 128 + kb + j];
        fv.v[j] = (bf16)Wv[(size_t)crow * 128 + kb + j];
      }
      accq = wmma_bf16(aS[ks], fq, accq);
      acck = wmma_bf16(aT[ks], fk, acck);
      accv = wmma_bf16(aT[ks], fv, accv);
    }
    const float biq = bq[crow], bik = bk[crow], biv = bv[crow];
    u16 vs[8];
#pragma unroll
    for (int r = 0; r < 8; ++r) {
      // D layout: lane column = crow, row voxel = n0 + r + 8*half
      const int n = n0 + r + 8 * half;
      Qb[((size_t)b * 4096 + n) * 128 + crow] = (bf16)((accq[r] + biq) * qscale);
      Kb[((size_t)b * 4096 + n) * 128 + crow] = (bf16)(acck[r] + bik);
      vs[r] = __builtin_bit_cast(u16, (bf16)(accv[r] + biv));
    }
    // Vt row crow: 8 consecutive voxels -> one 16B store
    uint4 vp;
    vp.x = (u32)vs[0] | ((u32)vs[1] << 16);
    vp.y = (u32)vs[2] | ((u32)vs[3] << 16);
    vp.z = (u32)vs[4] | ((u32)vs[5] << 16);
    vp.w = (u32)vs[6] | ((u32)vs[7] << 16);
    *(uint4*)(Vt + ((size_t)b * 128 + crow) * 4096 + n0 + 8 * half) = vp;
  }
}

// ---------------------------------------------------------------------------
// Kernel 2: flash attention. 4 waves, each owns a 16-row Q tile (WG = 64 rows).
// 64-col K/V tiles double-buffered in LDS via async global->LDS DMA.
// ---------------------------------------------------------------------------
__global__ __launch_bounds__(128) void attn_kernel(
    const bf16* __restrict__ Qb, const bf16* __restrict__ Kb,
    const bf16* __restrict__ Vt, bf16* __restrict__ Ob)
{
  __shared__ bf16 kT[2][64 * 128];   // 2 x 16 KB: K rows
  __shared__ bf16 vT[2][128 * 64];   // 2 x 16 KB: V^T rows
  __shared__ bf16 pT[4][16 * 64];    //     8 KB: wave-private P transpose

  const int b    = blockIdx.y;
  const int wave = threadIdx.x >> 5;
  const int lane = threadIdx.x & 31;
  const int m    = lane & 15;
  const int half = lane >> 4;
  const int tid  = threadIdx.x;
  const int row0 = blockIdx.x * 64 + wave * 16;

  // Q A-fragments live in registers for the whole kernel (32 VGPRs)
  Frag aQ[4];
  const bf16* qrow = Qb + ((size_t)b * 4096 + row0 + m) * 128;
#pragma unroll
  for (int ks = 0; ks < 4; ++ks) {
    aQ[ks].q[0] = *(const uint4*)(qrow + ks * 32 + half * 16);
    aQ[ks].q[1] = *(const uint4*)(qrow + ks * 32 + half * 16 + 8);
  }

  // stage one 64-col K/V tile into LDS buffer `buf`
  auto stage = [&](int j0, int buf) {
    const uint4* srcK = (const uint4*)(Kb + ((size_t)b * 4096 + j0) * 128);
    uint4* dstK = (uint4*)kT[buf];
    const uint4* srcV = (const uint4*)(Vt + ((size_t)b * 128 + tid) * 4096 + j0);
    uint4* dstV = (uint4*)(vT[buf] + tid * 64);
#if USE_ASYNC_LDS
#pragma unroll
    for (int i = 0; i < 8; ++i)
      __builtin_amdgcn_global_load_async_to_lds_b128(
          (gv4i*)(srcK + tid + i * 128), (lv4i*)(dstK + tid + i * 128), 0, 0);
#pragma unroll
    for (int i = 0; i < 8; ++i)
      __builtin_amdgcn_global_load_async_to_lds_b128(
          (gv4i*)(srcV + i), (lv4i*)(dstV + i), 0, 0);
#else
#pragma unroll
    for (int i = 0; i < 8; ++i) dstK[tid + i * 128] = srcK[tid + i * 128];
#pragma unroll
    for (int i = 0; i < 8; ++i) dstV[i] = srcV[i];
#endif
  };

  v8f zero = {};
  v8f O[8];
#pragma unroll
  for (int ct = 0; ct < 8; ++ct) O[ct] = zero;
  float rm[8], rl[8];
#pragma unroll
  for (int r = 0; r < 8; ++r) { rm[r] = -3.0e38f; rl[r] = 0.0f; }

  stage(0, 0);  // prologue: tile 0 in flight

  for (int it = 0; it < 64; ++it) {
    wait_async0();     // this wave's tile-`it` DMA has landed
    __syncthreads();   // => everyone's tile-`it` visible; prev reads of buf
                       //    (it+1)&1 (done in iter it-1) are globally complete
    if (it + 1 < 64) stage((it + 1) * 64, (it + 1) & 1);  // overlap with compute

    const bf16* kTb = kT[it & 1];
    const bf16* vTb = vT[it & 1];

    // S = Q * K^T : 16 x 64 per wave (4 tiles x 4 k-steps = 16 WMMA)
    v8f S[4];
#pragma unroll
    for (int jj = 0; jj < 4; ++jj) {
      v8f acc = zero;
#pragma unroll
      for (int ks = 0; ks < 4; ++ks) {
        Frag bK;  // B-layout: lane col = jj*16+m, 16 contiguous c
        const bf16* kr = kTb + (jj * 16 + m) * 128 + ks * 32 + half * 16;
        bK.q[0] = *(const uint4*)kr;
        bK.q[1] = *(const uint4*)(kr + 8);
        acc = wmma_bf16(aQ[ks], bK, acc);
      }
      S[jj] = acc;
    }

    // online softmax: row m = r + 8*half lives across a 16-lane half
#pragma unroll
    for (int r = 0; r < 8; ++r) {
      float mx = fmaxf(fmaxf(S[0][r], S[1][r]), fmaxf(S[2][r], S[3][r]));
#pragma unroll
      for (int off = 8; off >= 1; off >>= 1)
        mx = fmaxf(mx, __shfl_xor(mx, off, 32));
      const float nm   = fmaxf(rm[r], mx);
      const float resc = __expf(rm[r] - nm);
      rm[r] = nm;
      float ps = 0.0f;
#pragma unroll
      for (int jj = 0; jj < 4; ++jj) {
        const float p = __expf(S[jj][r] - nm);
        S[jj][r] = p;
        ps += p;
      }
#pragma unroll
      for (int off = 8; off >= 1; off >>= 1)
        ps += __shfl_xor(ps, off, 32);
      rl[r] = rl[r] * resc + ps;
#pragma unroll
      for (int ct = 0; ct < 8; ++ct) O[ct][r] *= resc;
    }

    // transpose P (D-layout -> A-layout) via wave-private LDS
    bf16* pw = pT[wave];
#pragma unroll
    for (int jj = 0; jj < 4; ++jj)
#pragma unroll
      for (int r = 0; r < 8; ++r)
        pw[(r + 8 * half) * 64 + jj * 16 + m] = (bf16)S[jj][r];

    // O += P * V : 8 c-tiles x 2 k-steps = 16 WMMA
#pragma unroll
    for (int ks2 = 0; ks2 < 2; ++ks2) {
      Frag aP;
      const bf16* pr = pw + m * 64 + ks2 * 32;
      aP.q[0] = *(const uint4*)(pr + half * 16);
      aP.q[1] = *(const uint4*)(pr + half * 16 + 8);
#pragma unroll
      for (int ct = 0; ct < 8; ++ct) {
        Frag bV;  // lane col = channel ct*16+m, 16 contiguous kv-cols
        const bf16* vr = vTb + (ct * 16 + m) * 64 + ks2 * 32 + half * 16;
        bV.q[0] = *(const uint4*)vr;
        bV.q[1] = *(const uint4*)(vr + 8);
        O[ct] = wmma_bf16(aP, bV, O[ct]);
      }
    }
  }

  // normalize and store attention output, bf16 [B][N][C]
#pragma unroll
  for (int ct = 0; ct < 8; ++ct)
#pragma unroll
    for (int r = 0; r < 8; ++r) {
      const int n = row0 + r + 8 * half;
      Ob[((size_t)b * 4096 + n) * 128 + ct * 16 + m] = (bf16)(O[ct][r] / rl[r]);
    }
}

// ---------------------------------------------------------------------------
// Kernel 3: output projection. Out[b, c_out, n] = Wo @ O + bo (f32 store).
// M = c_out, N-dim = voxel, K = c. A = Wo (contig), B = Ob [N][C] (contig).
// ---------------------------------------------------------------------------
__global__ __launch_bounds__(128) void out_proj_kernel(
    const bf16* __restrict__ Ob, const float* __restrict__ Wo,
    const float* __restrict__ bo, float* __restrict__ Out)
{
  const int b    = blockIdx.y;
  const int wave = threadIdx.x >> 5;
  const int lane = threadIdx.x & 31;
  const int m    = lane & 15;
  const int half = lane >> 4;
  const int n0   = blockIdx.x * 64 + wave * 16;

  Frag bO[4];
  const bf16* orow = Ob + ((size_t)b * 4096 + n0 + m) * 128;
#pragma unroll
  for (int ks = 0; ks < 4; ++ks) {
    bO[ks].q[0] = *(const uint4*)(orow + ks * 32 + half * 16);
    bO[ks].q[1] = *(const uint4*)(orow + ks * 32 + half * 16 + 8);
  }

#pragma unroll
  for (int ct = 0; ct < 8; ++ct) {
    v8f acc = {};
#pragma unroll
    for (int ks = 0; ks < 4; ++ks) {
      Frag aW;
      const float* w = Wo + (size_t)(ct * 16 + m) * 128 + ks * 32 + half * 16;
#pragma unroll
      for (int j = 0; j < 8; ++j) {
        aW.v[j]     = (bf16)w[j];
        aW.v[8 + j] = (bf16)w[8 + j];
      }
      acc = wmma_bf16(aW, bO[ks], acc);
    }
#pragma unroll
    for (int r = 0; r < 8; ++r) {
      const int cout = ct * 16 + r + 8 * half;
      Out[((size_t)b * 128 + cout) * 4096 + n0 + m] = acc[r] + bo[cout];
    }
  }
}

// ---------------------------------------------------------------------------
extern "C" void kernel_launch(void* const* d_in, const int* in_sizes, int n_in,
                              void* d_out, int out_size, void* d_ws, size_t ws_size,
                              hipStream_t stream) {
  (void)in_sizes; (void)n_in; (void)out_size; (void)ws_size;
  const float* spat = (const float*)d_in[0];
  const float* temp = (const float*)d_in[1];
  const float* Wq = (const float*)d_in[2];
  const float* bq = (const float*)d_in[3];
  const float* Wk = (const float*)d_in[4];
  const float* bk = (const float*)d_in[5];
  const float* Wv = (const float*)d_in[6];
  const float* bv = (const float*)d_in[7];
  const float* Wo = (const float*)d_in[8];
  const float* bo = (const float*)d_in[9];

  // Workspace partition: 4 bf16 buffers of B*N*C = 2,097,152 elements (4 MB each)
  const size_t seg = (size_t)4 * 4096 * 128;
  bf16* Qb = (bf16*)d_ws;
  bf16* Kb = Qb + seg;
  bf16* Vt = Kb + seg;
  bf16* Ob = Vt + seg;

  dim3 grid(64, 4), block(128);
  qkv_proj_kernel<<<grid, block, 0, stream>>>(spat, temp, Wq, bq, Wk, bk, Wv, bv,
                                              Qb, Kb, Vt);
  attn_kernel<<<grid, block, 0, stream>>>(Qb, Kb, Vt, Ob);
  out_proj_kernel<<<grid, block, 0, stream>>>(Ob, Wo, bo, (float*)d_out);
}
